// MultiHeadAttn_59115929862675
// MI455X (gfx1250) — matrix operand order repaired and checked by reading, not compiled, not run on hardware
//
#include <hip/hip_runtime.h>

typedef __bf16 bf16;
typedef __attribute__((ext_vector_type(16))) __bf16 v16bf;
typedef __attribute__((ext_vector_type(8)))  __bf16 v8bf;
typedef __attribute__((ext_vector_type(4)))  __bf16 v4bf;
typedef __attribute__((ext_vector_type(8)))  float  v8f;

#define B_    4
#define S_    2048
#define C_    1024
#define H_    16
#define D_    64
#define NQKV_ 3072
#define SCALE_INV 0.125f   // 1/sqrt(64)

#define WMMA_BF16(a, b, c) \
  __builtin_amdgcn_wmma_f32_16x16x32_bf16(false, (a), false, (b), (short)0, (c), false, false)

// ---- WMMA operand loaders (layouts per CDNA5 ISA 7.12.2) -------------------
// A operand (16x32, 16-bit): lane L holds row (L&15); elems 0..7 = K[hs*8 ..],
// elems 8..15 = K[16+hs*8 ..] where hs = L>>4.
static __device__ __forceinline__ v16bf load_a(const bf16* base, int ld, int lane) {
  int m  = lane & 15;
  int hs = lane >> 4;
  const bf16* p = base + (size_t)m * ld + hs * 8;
  v8bf lo = *(const v8bf*)(p);
  v8bf hi = *(const v8bf*)(p + 16);
  return __builtin_shufflevector(lo, hi, 0,1,2,3,4,5,6,7,8,9,10,11,12,13,14,15);
}
// B operand (32x16, 16-bit): lane L holds column (L&15); elems 0..7 = K[hs*8 ..],
// elems 8..15 = K[16+hs*8 ..] where hs = L>>4.  base = rows of B^T (K-contiguous).
static __device__ __forceinline__ v16bf load_b(const bf16* base, int ld, int lane) {
  int n  = lane & 15;
  int hs = lane >> 4;
  const bf16* p = base + (size_t)n * ld + hs * 8;
  v8bf lo = *(const v8bf*)(p);
  v8bf hi = *(const v8bf*)(p + 16);
  return __builtin_shufflevector(lo, hi, 0,1,2,3,4,5,6,7,8,9,10,11,12,13,14,15);
}

// ---- fp32 -> bf16 bulk convert --------------------------------------------
__global__ void __launch_bounds__(256) k_cvt_bf16(const float* __restrict__ src,
                                                  bf16* __restrict__ dst, int n) {
  int i = (blockIdx.x * 256 + threadIdx.x) * 4;
  if (i + 3 < n) {
    float4 v = *(const float4*)(src + i);
    v4bf o = {(bf16)v.x, (bf16)v.y, (bf16)v.z, (bf16)v.w};
    *(v4bf*)(dst + i) = o;
  }
}

// ---- GEMM1: qkv = x @ Wqkv^T + b, scattered to q[b,h,s,d], k[b,h,s,d], vT[b,h,d,s]
__global__ void __launch_bounds__(256) k_gemm_qkv(
    const bf16* __restrict__ Xb, const bf16* __restrict__ Wb,
    const float* __restrict__ bias,
    bf16* __restrict__ qb, bf16* __restrict__ kb, bf16* __restrict__ vT) {
  const int NSTRIP = NQKV_ / 64;   // 48
  int wid  = (blockIdx.x * 256 + threadIdx.x) >> 5;
  int lane = threadIdx.x & 31;
  int m0 = (wid / NSTRIP) * 16;
  int n0 = (wid % NSTRIP) * 64;

  v8f accs[4] = {};
  const bf16* arow = Xb + (size_t)m0 * C_;
  const bf16* brow[4] = {Wb + (size_t)(n0 + 0) * C_,  Wb + (size_t)(n0 + 16) * C_,
                         Wb + (size_t)(n0 + 32) * C_, Wb + (size_t)(n0 + 48) * C_};
  for (int k0 = 0; k0 < C_; k0 += 32) {
    __builtin_prefetch(arow + k0 + 128, 0, 1);
    v16bf a = load_a(arow + k0, C_, lane);
    #pragma unroll
    for (int t = 0; t < 4; ++t) {
      v16bf b = load_b(brow[t] + k0, C_, lane);
      accs[t] = WMMA_BF16(a, b, accs[t]);
    }
  }
  int hs = lane >> 4, nc = lane & 15;
  int mb    = m0 + hs * 8;
  int b_i   = mb >> 11;           // token / S   (tiles never cross a batch row)
  int sbase = mb & (S_ - 1);
  #pragma unroll
  for (int t = 0; t < 4; ++t) {
    int n = n0 + t * 16 + nc;
    float bv = bias[n];
    int sec = n >> 10;            // 0=q 1=k 2=v  (strip of 64 never crosses)
    int c = n & (C_ - 1);
    int h = c >> 6, d = c & 63;
    size_t bh = (size_t)b_i * H_ + h;
    if (sec == 2) {               // V transposed: s is contiguous -> packed store
      v8bf pk;
      #pragma unroll
      for (int r = 0; r < 8; ++r) pk[r] = (bf16)(accs[t][r] + bv);
      *(v8bf*)(vT + (bh * D_ + d) * S_ + sbase) = pk;
    } else {
      bf16* dst = (sec == 0 ? qb : kb) + (bh * S_ + sbase) * D_ + d;
      #pragma unroll
      for (int r = 0; r < 8; ++r) dst[(size_t)r * D_] = (bf16)(accs[t][r] + bv);
    }
  }
}

// ---- Streaming windowed attention: one wave per (b,h, 16-query tile) -------
__global__ void __launch_bounds__(32) k_attn(
    const bf16* __restrict__ qb, const bf16* __restrict__ kb,
    const bf16* __restrict__ vT, bf16* __restrict__ cb) {
  __shared__ bf16 pbuf[16 * 32];          // P tile restage (C-layout -> A-layout)
  int lane = threadIdx.x;
  int qt = blockIdx.x & (S_ / 16 - 1);    // 128 query tiles
  int bh = blockIdx.x >> 7;
  int s0 = qt * 16;

  const bf16* qbase = qb + ((size_t)bh * S_ + s0) * D_;
  const bf16* kbase = kb + (size_t)bh * S_ * D_;
  const bf16* vbase = vT + (size_t)bh * D_ * S_;

  v16bf qa0 = load_a(qbase,      D_, lane);   // d 0..31
  v16bf qa1 = load_a(qbase + 32, D_, lane);   // d 32..63

  v8f o[4] = {};
  float rmax[8], rsum[8];
  #pragma unroll
  for (int r = 0; r < 8; ++r) { rmax[r] = -1e30f; rsum[r] = 0.f; }

  int hs = lane >> 4, nc = lane & 15;
  int klo = s0 - 256; if (klo < 0) klo = 0;

  for (int j0 = klo; j0 <= s0 + 15; j0 += 32) {
    // ---- scores: 16 queries x 32 keys (two 16x16 halves), K-dim = d (2 steps)
    v8f sa0 = {}, sa1 = {};
    {
      v16bf kb00 = load_b(kbase + (size_t)j0 * D_,             D_, lane);
      v16bf kb01 = load_b(kbase + (size_t)j0 * D_ + 32,        D_, lane);
      v16bf kb10 = load_b(kbase + (size_t)(j0 + 16) * D_,      D_, lane);
      v16bf kb11 = load_b(kbase + (size_t)(j0 + 16) * D_ + 32, D_, lane);
      sa0 = WMMA_BF16(qa0, kb00, sa0);
      sa0 = WMMA_BF16(qa1, kb01, sa0);
      sa1 = WMMA_BF16(qa0, kb10, sa1);
      sa1 = WMMA_BF16(qa1, kb11, sa1);
    }
    __syncthreads();                       // protect pbuf vs previous read
    #pragma unroll
    for (int r = 0; r < 8; ++r) {
      int qi  = s0 + hs * 8 + r;
      int kj0 = j0 + nc;
      int kj1 = kj0 + 16;
      float x0 = sa0[r] * SCALE_INV;
      float x1 = sa1[r] * SCALE_INV;
      x0 = ((kj0 <= qi) && (qi - kj0 < 256)) ? x0 : -1e30f;
      x1 = ((kj1 <= qi) && (qi - kj1 < 256)) ? x1 : -1e30f;
      // row max across 16 lanes (butterfly within half-wave group)
      float mx = fmaxf(x0, x1);
      mx = fmaxf(mx, __shfl_xor(mx, 1, 32));
      mx = fmaxf(mx, __shfl_xor(mx, 2, 32));
      mx = fmaxf(mx, __shfl_xor(mx, 4, 32));
      mx = fmaxf(mx, __shfl_xor(mx, 8, 32));
      float nm = fmaxf(rmax[r], mx);
      float sf = __expf(rmax[r] - nm);
      rmax[r] = nm;
      float p0 = __expf(x0 - nm);
      float p1 = __expf(x1 - nm);
      float ps = p0 + p1;
      ps += __shfl_xor(ps, 1, 32);
      ps += __shfl_xor(ps, 2, 32);
      ps += __shfl_xor(ps, 4, 32);
      ps += __shfl_xor(ps, 8, 32);
      rsum[r] = rsum[r] * sf + ps;
      int row = hs * 8 + r;
      pbuf[row * 32 + nc]      = (bf16)p0;
      pbuf[row * 32 + 16 + nc] = (bf16)p1;
      o[0][r] *= sf; o[1][r] *= sf; o[2][r] *= sf; o[3][r] *= sf;
    }
    __syncthreads();
    // ---- P (A operand, 16x32) from LDS; V^T (B operand, keys contiguous)
    v16bf pa;
    {
      int m = lane & 15;
      v8bf lo = *(const v8bf*)&pbuf[m * 32 + hs * 8];
      v8bf hi = *(const v8bf*)&pbuf[m * 32 + 16 + hs * 8];
      pa = __builtin_shufflevector(lo, hi, 0,1,2,3,4,5,6,7,8,9,10,11,12,13,14,15);
    }
    #pragma unroll
    for (int t = 0; t < 4; ++t) {
      v16bf vb = load_b(vbase + (size_t)(t * 16) * S_ + j0, S_, lane);
      o[t] = WMMA_BF16(pa, vb, o[t]);
    }
  }
  // ---- normalize + merge heads into ctx[b,s,h*64+d] (bf16)
  int b_i = bh >> 4;   // / H_
  int h   = bh & (H_ - 1);
  float inv[8];
  #pragma unroll
  for (int r = 0; r < 8; ++r) inv[r] = 1.0f / rsum[r];
  #pragma unroll
  for (int t = 0; t < 4; ++t) {
    int d = h * 64 + t * 16 + nc;
    #pragma unroll
    for (int r = 0; r < 8; ++r) {
      int s = s0 + hs * 8 + r;
      cb[((size_t)b_i * S_ + s) * C_ + d] = (bf16)(o[t][r] * inv[r]);
    }
  }
}

// ---- GEMM2: out = ctx @ Wproj^T + b (fp32 out) -----------------------------
__global__ void __launch_bounds__(256) k_gemm_proj(
    const bf16* __restrict__ Cb, const bf16* __restrict__ Wp,
    const float* __restrict__ bias, float* __restrict__ out) {
  const int NSTRIP = C_ / 64;   // 16
  int wid  = (blockIdx.x * 256 + threadIdx.x) >> 5;
  int lane = threadIdx.x & 31;
  int m0 = (wid / NSTRIP) * 16;
  int n0 = (wid % NSTRIP) * 64;

  v8f accs[4] = {};
  const bf16* arow = Cb + (size_t)m0 * C_;
  const bf16* brow[4] = {Wp + (size_t)(n0 + 0) * C_,  Wp + (size_t)(n0 + 16) * C_,
                         Wp + (size_t)(n0 + 32) * C_, Wp + (size_t)(n0 + 48) * C_};
  for (int k0 = 0; k0 < C_; k0 += 32) {
    __builtin_prefetch(arow + k0 + 128, 0, 1);
    v16bf a = load_a(arow + k0, C_, lane);
    #pragma unroll
    for (int t = 0; t < 4; ++t) {
      v16bf b = load_b(brow[t] + k0, C_, lane);
      accs[t] = WMMA_BF16(a, b, accs[t]);
    }
  }
  int hs = lane >> 4, nc = lane & 15;
  #pragma unroll
  for (int t = 0; t < 4; ++t) {
    int n = n0 + t * 16 + nc;
    float bv = bias[n];
    #pragma unroll
    for (int r = 0; r < 8; ++r) {
      size_t tok = (size_t)m0 + hs * 8 + r;
      out[tok * C_ + n] = accs[t][r] + bv;
    }
  }
}

extern "C" void kernel_launch(void* const* d_in, const int* in_sizes, int n_in,
                              void* d_out, int out_size, void* d_ws, size_t ws_size,
                              hipStream_t stream) {
  (void)in_sizes; (void)n_in; (void)out_size; (void)ws_size;
  const float* x     = (const float*)d_in[0];
  const float* Wqkv  = (const float*)d_in[1];
  const float* bqkv  = (const float*)d_in[2];
  const float* Wproj = (const float*)d_in[3];
  const float* bproj = (const float*)d_in[4];
  float* out = (float*)d_out;

  // workspace carve-out (~88 MB), 256B aligned
  size_t off = 0;
  char* base = (char*)d_ws;
  auto carve = [&](size_t bytes) { char* p = base + off; off += (bytes + 255) & ~(size_t)255; return p; };
  bf16* xb     = (bf16*)carve((size_t)B_ * S_ * C_ * 2);     // 16 MB
  bf16* wqkvb  = (bf16*)carve((size_t)NQKV_ * C_ * 2);       //  6 MB
  bf16* wprojb = (bf16*)carve((size_t)C_ * C_ * 2);          //  2 MB
  bf16* qbuf   = (bf16*)carve((size_t)B_ * H_ * S_ * D_ * 2);// 16 MB
  bf16* kbuf   = (bf16*)carve((size_t)B_ * H_ * S_ * D_ * 2);// 16 MB (tail OOB reads land in vT: masked)
  bf16* vTbuf  = (bf16*)carve((size_t)B_ * H_ * D_ * S_ * 2);// 16 MB (tail OOB reads land in cb: masked)
  bf16* cbuf   = (bf16*)carve((size_t)B_ * S_ * C_ * 2);     // 16 MB

  // 1) fp32 -> bf16 converts
  k_cvt_bf16<<<(B_ * S_ * C_) / 1024,  256, 0, stream>>>(x,     xb,     B_ * S_ * C_);
  k_cvt_bf16<<<(NQKV_ * C_) / 1024,    256, 0, stream>>>(Wqkv,  wqkvb,  NQKV_ * C_);
  k_cvt_bf16<<<(C_ * C_) / 1024,       256, 0, stream>>>(Wproj, wprojb, C_ * C_);

  // 2) QKV projection: waves = (8192/16)*(3072/64) = 24576 -> 3072 blocks x 256
  k_gemm_qkv<<<3072, 256, 0, stream>>>(xb, wqkvb, bqkv, qbuf, kbuf, vTbuf);

  // 3) windowed attention: one wave per (b,h,qtile) = 8192 blocks x 32
  k_attn<<<B_ * H_ * (S_ / 16), 32, 0, stream>>>(qbuf, kbuf, vTbuf, cbuf);

  // 4) output projection: waves = (8192/16)*(1024/64) = 8192 -> 1024 blocks x 256
  k_gemm_proj<<<1024, 256, 0, stream>>>(cbuf, wprojb, bproj, out);
}
